// Matposer_51007031608225
// MI455X (gfx1250) — matrix operand order repaired and checked by reading, not compiled, not run on hardware
//
#include <hip/hip_runtime.h>
#include <hip/hip_bf16.h>
#include <math.h>

// Problem constants from the reference
#define D_MODEL 300
#define LSEQ    512
#define OUT_N   4
#define MPAD    304            // M (e rows) padded to 16-multiple
#define KPAD    320            // K (i cols) padded to 32-multiple
#define NPAD    304            // N (w1 rows j) padded to 16-multiple
#define PACK_ELEMS (MPAD * KPAD)   // == KPAD * NPAD == 97280
#define SCALE   17.3205080757f               // sqrt(300)
#define LN1E4_OVER_D 0.03070113457325395f    // ln(10000)/300

typedef __attribute__((ext_vector_type(16))) _Float16 v16h;
typedef __attribute__((ext_vector_type(8)))  _Float16 v8h;
typedef __attribute__((ext_vector_type(8)))  float    v8f;

// -----------------------------------------------------------------------------
// Prep: zero-fill padded f16 A buffer; build padded f16 B = w1^T.
//   Apack[m][k] : [MPAD][KPAD]  (diag_kernel fills the valid region after this)
//   Bpack[k][n] : [KPAD][NPAD]  = w1[n][k], zeros in padding
// -----------------------------------------------------------------------------
__global__ __launch_bounds__(256) void prep_kernel(
    const float* __restrict__ w1,
    _Float16*    __restrict__ Apack,
    _Float16*    __restrict__ Bpack)
{
    const int idx = blockIdx.x * 256 + threadIdx.x;
    if (idx < PACK_ELEMS) {
        Apack[idx] = (_Float16)0.0f;
        const int k = idx / NPAD;
        const int n = idx - k * NPAD;
        float v = 0.f;
        if (k < D_MODEL && n < D_MODEL) v = w1[n * D_MODEL + k];
        Bpack[idx] = (_Float16)v;
    }
}

// -----------------------------------------------------------------------------
// Stage 1: diag[e][i] = sum_l (emb2[x2[i,l]][i]*s + pe(l,i)) *
//                             (emb1[x1[i,l]][e]*s + pe(l,e))
// One block per i (300 blocks), 320 threads (10 wave32).
// Result written as f16 directly into the padded WMMA A buffer.
// -----------------------------------------------------------------------------
__global__ __launch_bounds__(320) void diag_kernel(
    const int*   __restrict__ x1,
    const int*   __restrict__ x2,
    const float* __restrict__ emb1,
    const float* __restrict__ emb2,
    _Float16*    __restrict__ Apack)   // [MPAD][KPAD], row = e, col = i
{
    __shared__ float s[LSEQ];
    const int i   = blockIdx.x;        // 0..299  (batch index == d index)
    const int tid = threadIdx.x;       // 0..319

    // pe(l, d) = (d even) ? sin(l*f) : cos(l*f), f = exp(-(d&~1)*ln(1e4)/300)
    {
        const float freq_i = __expf(-LN1E4_OVER_D * (float)(i & ~1));
        const bool  odd_i  = (i & 1);
        for (int l = tid; l < LSEQ; l += 320) {
            const int tok = x2[i * LSEQ + l];
            const float pev = odd_i ? __cosf((float)l * freq_i)
                                    : __sinf((float)l * freq_i);
            s[l] = emb2[tok * D_MODEL + i] * SCALE + pev;
        }
    }
    __syncthreads();

    if (tid < D_MODEL) {
        const int   e      = tid;
        const float freq_e = __expf(-LN1E4_OVER_D * (float)(e & ~1));
        const bool  odd_e  = (e & 1);
        float acc = 0.f;
        for (int l = 0; l < LSEQ; ++l) {
            const int tok = x1[i * LSEQ + l];              // uniform -> scalar load
            const float pev = odd_e ? __cosf((float)l * freq_e)
                                    : __sinf((float)l * freq_e);
            const float a = emb1[tok * D_MODEL + e] * SCALE + pev; // coalesced
            acc = fmaf(s[l], a, acc);
        }
        Apack[e * KPAD + i] = (_Float16)acc;
    }
}

// -----------------------------------------------------------------------------
// Stage 2+3: h = relu(diag @ w1^T + b1) via v_wmma_f32_16x16x32_f16,
// then out = softmax(h @ w2^T + b2) per row.
// One wave32 block per M-tile of 16 rows; 19 blocks.
// Fragment layouts (ISA 7.12.2, wave32):
//   A 16x32 f16 : lane L<16 -> M=L, halves 0..7 = K 0..7,  halves 8..15 = K 16..23
//                 lane L>=16 -> M=L-16, halves = K 8..15 / 24..31
//                 => two contiguous 8xf16 (16B) loads per lane.
//   B 32x16 f16 : lane L -> N=L%16, halves = K in the same order as A (16 strided loads).
//   C/D 16x16 f32: lane%16 = N, vgpr v = M row (lanes>=16 -> M=8+v).
// -----------------------------------------------------------------------------
__global__ __launch_bounds__(32) void fc_softmax_kernel(
    const _Float16* __restrict__ Apack,  // [MPAD][KPAD]
    const _Float16* __restrict__ Bpack,  // [KPAD][NPAD]
    const float*    __restrict__ b1,     // [300]
    const float*    __restrict__ w2,     // [4][300]
    const float*    __restrict__ b2,     // [4]
    float*          __restrict__ out)    // [300][4]
{
    __shared__ float h_lds[16 * NPAD];

    const int m     = blockIdx.x;            // M tile (0..18)
    const int lane  = threadIdx.x;           // 0..31
    const int Mrow  = m * 16 + (lane & 15);  // row of A for this lane (<= 303, padded)
    const int kHalf = (lane < 16) ? 0 : 8;   // A: K sub-offset per lane half
    const int mBase = (lane < 16) ? 0 : 8;   // D: M row base per lane half

    const _Float16* arow = Apack + (size_t)Mrow * KPAD;

    for (int nt = 0; nt < 19; ++nt) {          // N tiles of 16
        v8f acc = {};
        const _Float16* bcol = Bpack + nt * 16 + (lane & 15);
        for (int kc = 0; kc < 10; ++kc) {      // K chunks of 32
            const int kBase = kc * 32;

            const v8h alo = *(const v8h*)(arow + kBase + kHalf);
            const v8h ahi = *(const v8h*)(arow + kBase + 16 + kHalf);
            const v16h a = __builtin_shufflevector(alo, ahi,
                0,1,2,3,4,5,6,7,8,9,10,11,12,13,14,15);

            const _Float16* brow = bcol + (size_t)kBase * NPAD;
            v16h b;
            #pragma unroll
            for (int s = 0; s < 16; ++s)
                b[s] = brow[(size_t)(kHalf + (s < 8 ? s : s + 8)) * NPAD];

            acc = __builtin_amdgcn_wmma_f32_16x16x32_f16(
                false, a, false, b, (short)0, acc, false, false);
        }

        // bias + relu, stash tile in LDS
        const int ncol = nt * 16 + (lane & 15);
        const float bias = (ncol < D_MODEL) ? b1[ncol] : 0.f;
        #pragma unroll
        for (int v = 0; v < 8; ++v) {
            float hv = acc[v] + bias;
            hv = fmaxf(hv, 0.f);
            h_lds[(mBase + v) * NPAD + ncol] = hv;
        }
    }
    __syncthreads();

    // Stage 3: logits + softmax; one lane per output row.
    if (lane < 16) {
        const int row = m * 16 + lane;
        if (row < D_MODEL) {
            float lg[OUT_N];
            #pragma unroll
            for (int o = 0; o < OUT_N; ++o) {
                float acc = b2[o];
                for (int j = 0; j < D_MODEL; ++j)
                    acc = fmaf(h_lds[lane * NPAD + j], w2[o * D_MODEL + j], acc);
                lg[o] = acc;
            }
            float mx = fmaxf(fmaxf(lg[0], lg[1]), fmaxf(lg[2], lg[3]));
            float ex[OUT_N], sum = 0.f;
            #pragma unroll
            for (int o = 0; o < OUT_N; ++o) { ex[o] = __expf(lg[o] - mx); sum += ex[o]; }
            const float inv = 1.f / sum;
            #pragma unroll
            for (int o = 0; o < OUT_N; ++o)
                out[row * OUT_N + o] = ex[o] * inv;
        }
    }
}

extern "C" void kernel_launch(void* const* d_in, const int* in_sizes, int n_in,
                              void* d_out, int out_size, void* d_ws, size_t ws_size,
                              hipStream_t stream) {
    (void)in_sizes; (void)n_in; (void)out_size; (void)ws_size;
    const int*   x1   = (const int*)  d_in[0];
    const int*   x2   = (const int*)  d_in[1];
    const float* emb1 = (const float*)d_in[2];
    const float* emb2 = (const float*)d_in[3];
    const float* w1   = (const float*)d_in[4];
    const float* b1   = (const float*)d_in[5];
    const float* w2   = (const float*)d_in[6];
    const float* b2   = (const float*)d_in[7];
    float* out = (float*)d_out;

    _Float16* Apack = (_Float16*)d_ws;                 // 194,560 B
    _Float16* Bpack = Apack + PACK_ELEMS;              // 194,560 B (total ~380 KB)

    prep_kernel<<<(PACK_ELEMS + 255) / 256, 256, 0, stream>>>(w1, Apack, Bpack);
    diag_kernel<<<D_MODEL, 320, 0, stream>>>(x1, x2, emb1, emb2, Apack);
    fc_softmax_kernel<<<19, 32, 0, stream>>>(Apack, Bpack, b1, w2, b2, out);
}
